// DecoderWithAttention_7980049236036
// MI455X (gfx1250) — matrix operand order repaired and checked by reading, not compiled, not run on hardware
//
#include <hip/hip_runtime.h>

// ---------------------------------------------------------------------------
// Show-Attend-Tell decoder for MI455X (gfx1250).
// Strategy: weights/activations converted to bf16 once, all GEMMs run on
// v_wmma_f32_16x16x32_bf16 (f32 accumulate). The whole working set (~150MB)
// fits in the 192MB L2, so the sequential t-loop is L2-BW bound; bf16 halves
// per-step traffic. W_ih and W_hh are fused into one [2048 x 3072] matrix so
// the LSTM input GEMM is a single pass.
// Round-1 fix: prefetch locality 0 lowered to SCOPE_SYS (GL2-only), useless
// for an L2-resident working set. Use locality 3 -> WGP-scope prefetch that
// pulls the K-stream into the WGP cache ahead of the b128 loads.
// ---------------------------------------------------------------------------

#define BB   64
#define PP   196
#define DENCC 2048
#define AAtt 512
#define EEmb 512
#define DHid 512
#define VVoc 10000
#define MLen 50
#define TSt  49
#define XKc  3072           // E + DENC + D (fused LSTM input)

typedef __bf16 bf16_t;
typedef bf16_t v16bf __attribute__((ext_vector_type(16)));
typedef float  v8f   __attribute__((ext_vector_type(8)));
typedef unsigned int u32x4 __attribute__((ext_vector_type(4)));

__device__ __forceinline__ unsigned short f2bf(float f) {
  unsigned int u = __float_as_uint(f);
  u += 0x7FFFu + ((u >> 16) & 1u);          // round-to-nearest-even
  return (unsigned short)(u >> 16);
}
__device__ __forceinline__ float bf2f(unsigned short b) {
  return __uint_as_float(((unsigned int)b) << 16);
}
__device__ __forceinline__ float sigm(float x) { return 1.0f / (1.0f + __expf(-x)); }

// ---------------------------------------------------------------------------
// Generic bf16 WMMA GEMM:  D[M,N] = X[M,K] * W[N,K]^T + bias   (row-major)
// One wave computes one 16x16 tile; K streamed in chunks of 32.
// A-frag per lane (half=lane>>4, l=lane&15): X[m0+l][kk+8*half .. +7] and
//   X[m0+l][kk+16+8*half .. +7]     (matches 16-bit 16x32 A layout)
// B-frag per lane: W[n0+l][kk+8*half .. +7] and W[n0+l][kk+16+8*half .. +7]
// mode 2: zero rows where decode_lengths[m] <= t  (prediction masking)
// ---------------------------------------------------------------------------
__global__ void k_gemm_bf16(const bf16_t* __restrict__ X, const bf16_t* __restrict__ W,
                            const float* __restrict__ bias, float* __restrict__ Dm,
                            int M, int N, int K, long ldd,
                            int mode, const int* __restrict__ declen, int t) {
  int wave  = blockIdx.x * (blockDim.x >> 5) + (threadIdx.x >> 5);
  int ntN   = N >> 4;
  int total = (M >> 4) * ntN;
  if (wave >= total) return;                 // wave-uniform: EXEC stays all-ones
  int m0 = (wave / ntN) << 4;
  int n0 = (wave % ntN) << 4;
  int lane = threadIdx.x & 31;
  int half = lane >> 4;
  int l    = lane & 15;
  const bf16_t* xr = X + (long)(m0 + l) * K;
  const bf16_t* wr = W + (long)(n0 + l) * K;
  v8f acc = {0.f, 0.f, 0.f, 0.f, 0.f, 0.f, 0.f, 0.f};
  for (int kk = 0; kk < K; kk += 32) {
    union Frag { v16bf v; u32x4 u[2]; } af, bfg;
    af.u[0]  = *(const u32x4*)(xr + kk + half * 8);
    af.u[1]  = *(const u32x4*)(xr + kk + 16 + half * 8);
    bfg.u[0] = *(const u32x4*)(wr + kk + half * 8);
    bfg.u[1] = *(const u32x4*)(wr + kk + 16 + half * 8);
    // WGP-scope prefetch (locality 3): pull next K-chunks into WGP$/L0.
    __builtin_prefetch(xr + kk + 128, 0, 3);
    __builtin_prefetch(wr + kk + 128, 0, 3);
    acc = __builtin_amdgcn_wmma_f32_16x16x32_bf16(
        false, af.v, false, bfg.v, (short)0, acc, false, false);
  }
  int   n  = n0 + l;
  float bv = bias ? bias[n] : 0.0f;
#pragma unroll
  for (int r = 0; r < 8; ++r) {
    int   m = m0 + r + half * 8;
    float v = acc[r] + bv;
    if (mode == 2) v *= (declen[m] > t) ? 1.0f : 0.0f;
    Dm[(long)m * ldd + n] = v;
  }
}

// ---------------------------------------------------------------------------
// Stable descending sort by caption length (B=64, one block) + outputs
// ---------------------------------------------------------------------------
__global__ void k_sort(const int* __restrict__ lengths, const int* __restrict__ caps,
                       int* __restrict__ order, int* __restrict__ declen,
                       float* __restrict__ out_caps, float* __restrict__ out_declen,
                       float* __restrict__ out_order) {
  __shared__ int ord[BB];
  int i  = threadIdx.x;
  int li = lengths[i];
  int rank = 0;
  for (int j = 0; j < BB; ++j) {
    int lj = lengths[j];
    if (lj > li || (lj == li && j < i)) rank++;
  }
  ord[rank] = i;
  __syncthreads();
  int src = ord[i];
  order[i] = src;
  int dl = lengths[src] - 1;
  declen[i]     = dl;
  out_declen[i] = (float)dl;
  out_order[i]  = (float)src;
  for (int k = 0; k < MLen; ++k)
    out_caps[i * MLen + k] = (float)caps[src * MLen + k];
}

// gather sorted encoder features and convert to bf16
__global__ void k_enc_gather(const float* __restrict__ enc, const int* __restrict__ order,
                             unsigned short* __restrict__ dst) {
  const long n = (long)BB * PP * DENCC;
  long stride = (long)gridDim.x * blockDim.x;
  for (long i = (long)blockIdx.x * blockDim.x + threadIdx.x; i < n; i += stride) {
    int  b   = (int)(i / ((long)PP * DENCC));
    long rem = i % ((long)PP * DENCC);
    dst[i] = f2bf(enc[(long)order[b] * PP * DENCC + rem]);
  }
}

// mean over P (f32 source for precision), output bf16 for the init GEMMs
__global__ void k_mean(const float* __restrict__ enc, const int* __restrict__ order,
                       unsigned short* __restrict__ mean_bf) {
  int b = blockIdx.x;
  const float* e = enc + (long)order[b] * PP * DENCC;
  for (int d = threadIdx.x; d < DENCC; d += blockDim.x) {
    float s = 0.f;
    for (int p = 0; p < PP; ++p) s += e[(long)p * DENCC + d];
    mean_bf[(long)b * DENCC + d] = f2bf(s * (1.0f / PP));
  }
}

// generic f32 -> bf16 convert
__global__ void k_cvt(const float* __restrict__ src, unsigned short* __restrict__ dst, long n) {
  long stride = (long)gridDim.x * blockDim.x;
  for (long i = (long)blockIdx.x * blockDim.x + threadIdx.x; i < n; i += stride)
    dst[i] = f2bf(src[i]);
}

// fuse W_ih [2048x2560] and W_hh [2048x512] -> W_cat [2048x3072] bf16
__global__ void k_wcat(const float* __restrict__ Wih, const float* __restrict__ Whh,
                       unsigned short* __restrict__ Wcat) {
  const long n = (long)4 * DHid * XKc;
  long stride = (long)gridDim.x * blockDim.x;
  for (long i = (long)blockIdx.x * blockDim.x + threadIdx.x; i < n; i += stride) {
    int row = (int)(i / XKc);
    int j   = (int)(i % XKc);
    float v = (j < EEmb + DENCC) ? Wih[(long)row * (EEmb + DENCC) + j]
                                 : Whh[(long)row * DHid + (j - EEmb - DENCC)];
    Wcat[i] = f2bf(v);
  }
}

__global__ void k_bcat(const float* __restrict__ bih, const float* __restrict__ bhh,
                       float* __restrict__ bcat) {
  int i = blockIdx.x * blockDim.x + threadIdx.x;
  if (i < 4 * DHid) bcat[i] = bih[i] + bhh[i];
}

// att[b,p] = b_full + sum_a relu(enc_proj[b,p,a]+dec_proj[b,a]) * w_full[a]
__global__ void k_att(const float* __restrict__ enc_proj, const float* __restrict__ dec_proj,
                      const float* __restrict__ w_full, const float* __restrict__ b_full,
                      float* __restrict__ att) {
  int bp = blockIdx.x * (blockDim.x >> 5) + (threadIdx.x >> 5);
  if (bp >= BB * PP) return;
  int b    = bp / PP;
  int lane = threadIdx.x & 31;
  const float* ep = enc_proj + (long)bp * AAtt;
  const float* dp = dec_proj + (long)b * AAtt;
  float s = 0.f;
  for (int a = lane; a < AAtt; a += 32) {
    float v = ep[a] + dp[a];
    s += (v > 0.f ? v : 0.f) * w_full[a];
  }
  for (int off = 16; off; off >>= 1) s += __shfl_xor(s, off, 32);
  if (lane == 0) att[bp] = s + b_full[0];
}

// softmax over P=196 per row; writes alpha (ws) and masked alphas (output)
__global__ void k_softmax(const float* __restrict__ att, float* __restrict__ alpha,
                          float* __restrict__ alphas_out, const int* __restrict__ declen,
                          int t) {
  __shared__ float red[8];
  int b = blockIdx.x, tid = threadIdx.x;
  float m = -1e30f;
  for (int p = tid; p < PP; p += 256) m = fmaxf(m, att[b * PP + p]);
  for (int off = 16; off; off >>= 1) m = fmaxf(m, __shfl_xor(m, off, 32));
  if ((tid & 31) == 0) red[tid >> 5] = m;
  __syncthreads();
  m = red[0];
  for (int w = 1; w < 8; ++w) m = fmaxf(m, red[w]);
  __syncthreads();
  float s = 0.f;
  for (int p = tid; p < PP; p += 256) s += __expf(att[b * PP + p] - m);
  for (int off = 16; off; off >>= 1) s += __shfl_xor(s, off, 32);
  if ((tid & 31) == 0) red[tid >> 5] = s;
  __syncthreads();
  s = 0.f;
  for (int w = 0; w < 8; ++w) s += red[w];
  float inv = 1.0f / s;
  float act = (declen[b] > t) ? 1.0f : 0.0f;
  for (int p = tid; p < PP; p += 256) {
    float al = __expf(att[b * PP + p] - m) * inv;
    alpha[b * PP + p] = al;
    alphas_out[((long)b * TSt + t) * PP + p] = al * act;
  }
}

// awe[b,d] = sum_p alpha[b,p] * enc_bf16[b,p,d]   (alpha cached in LDS)
__global__ void k_awe(const float* __restrict__ alpha, const unsigned short* __restrict__ encbf,
                      float* __restrict__ awe) {
  __shared__ float al[PP];
  int b = blockIdx.x, tid = threadIdx.x;
  for (int p = tid; p < PP; p += blockDim.x) al[p] = alpha[b * PP + p];
  __syncthreads();
  const unsigned short* e = encbf + (long)b * PP * DENCC;
  for (int d = tid; d < DENCC; d += blockDim.x) {
    float s = 0.f;
    for (int p = 0; p < PP; ++p) s += al[p] * bf2f(e[(long)p * DENCC + d]);
    awe[(long)b * DENCC + d] = s;
  }
}

// x[b] = [ emb(cap_t) , sigmoid(gatepre)*awe , h ]  in bf16
__global__ void k_buildx(const int* __restrict__ caps, const int* __restrict__ order,
                         const float* __restrict__ emb_table, const float* __restrict__ gatepre,
                         const float* __restrict__ awe, const float* __restrict__ h,
                         unsigned short* __restrict__ x, int t) {
  int b   = blockIdx.x;
  int cap = caps[order[b] * MLen + t];
  for (int j = threadIdx.x; j < XKc; j += blockDim.x) {
    float v;
    if (j < EEmb) {
      v = emb_table[(long)cap * EEmb + j];
    } else if (j < EEmb + DENCC) {
      int d = j - EEmb;
      v = sigm(gatepre[(long)b * DENCC + d]) * awe[(long)b * DENCC + d];
    } else {
      v = h[(long)b * DHid + (j - EEmb - DENCC)];
    }
    x[(long)b * XKc + j] = f2bf(v);
  }
}

// LSTM pointwise (PyTorch gate order i,f,g,o) + active masking of h/c
__global__ void k_lstm(const float* __restrict__ gates, float* __restrict__ h,
                       float* __restrict__ c, unsigned short* __restrict__ h_bf,
                       unsigned short* __restrict__ hnew_bf, const int* __restrict__ declen,
                       int t) {
  int idx = blockIdx.x * blockDim.x + threadIdx.x;
  if (idx >= BB * DHid) return;
  int b = idx >> 9, n = idx & (DHid - 1);
  const float* g = gates + (long)b * 4 * DHid;
  float gi = sigm(g[n]);
  float gf = sigm(g[DHid + n]);
  float gg = tanhf(g[2 * DHid + n]);
  float go = sigm(g[3 * DHid + n]);
  float cn = gf * c[idx] + gi * gg;
  float hn = go * tanhf(cn);
  bool  act  = declen[b] > t;
  float hcur = act ? hn : h[idx];
  float ccur = act ? cn : c[idx];
  h[idx] = hcur;
  c[idx] = ccur;
  h_bf[idx]    = f2bf(hcur);
  hnew_bf[idx] = f2bf(hn);
}

// ---------------------------------------------------------------------------
extern "C" void kernel_launch(void* const* d_in, const int* in_sizes, int n_in,
                              void* d_out, int out_size, void* d_ws, size_t ws_size,
                              hipStream_t stream) {
  const float* encoder_out = (const float*)d_in[0];
  const int*   enc_caps    = (const int*)d_in[1];
  const int*   cap_len     = (const int*)d_in[2];
  const float* emb_table   = (const float*)d_in[3];
  const float* W_enc_att   = (const float*)d_in[4];
  const float* b_enc_att   = (const float*)d_in[5];
  const float* W_dec_att   = (const float*)d_in[6];
  const float* b_dec_att   = (const float*)d_in[7];
  const float* w_full      = (const float*)d_in[8];
  const float* b_full      = (const float*)d_in[9];
  const float* W_init_h    = (const float*)d_in[10];
  const float* b_init_h    = (const float*)d_in[11];
  const float* W_init_c    = (const float*)d_in[12];
  const float* b_init_c    = (const float*)d_in[13];
  const float* W_f_beta    = (const float*)d_in[14];
  const float* b_f_beta    = (const float*)d_in[15];
  const float* W_ih        = (const float*)d_in[16];
  const float* b_ih        = (const float*)d_in[17];
  const float* W_hh        = (const float*)d_in[18];
  const float* b_hh        = (const float*)d_in[19];
  const float* W_fc        = (const float*)d_in[20];
  const float* b_fc        = (const float*)d_in[21];

  // output layout (flat, return order): preds, caps, decode_lengths, alphas, order
  float* out        = (float*)d_out;
  float* out_preds  = out;                                   // 64*49*10000
  float* out_caps   = out + (long)BB * TSt * VVoc;           // 64*50
  float* out_declen = out_caps + BB * MLen;                  // 64
  float* out_alphas = out_declen + BB;                       // 64*49*196
  float* out_order  = out_alphas + (long)BB * TSt * PP;      // 64

  // workspace carve-up (256B aligned)
  size_t off = 0;
  auto alloc = [&](size_t bytes) -> void* {
    void* p = (char*)d_ws + off;
    off = (off + bytes + 255) & ~(size_t)255;
    return p;
  };
  int*            ws_order  = (int*)alloc(BB * 4);
  int*            ws_declen = (int*)alloc(BB * 4);
  unsigned short* enc_bf    = (unsigned short*)alloc((long)BB * PP * DENCC * 2);
  float*          enc_proj  = (float*)alloc((long)BB * PP * AAtt * 4);
  unsigned short* Wenc_bf   = (unsigned short*)alloc((long)AAtt * DENCC * 2);
  unsigned short* Wdec_bf   = (unsigned short*)alloc((long)AAtt * DHid * 2);
  unsigned short* Wfb_bf    = (unsigned short*)alloc((long)DENCC * DHid * 2);
  unsigned short* Wcat_bf   = (unsigned short*)alloc((long)4 * DHid * XKc * 2);
  unsigned short* Wfc_bf    = (unsigned short*)alloc((long)VVoc * DHid * 2);
  unsigned short* Wih0_bf   = (unsigned short*)alloc((long)DHid * DENCC * 2);
  unsigned short* Wic0_bf   = (unsigned short*)alloc((long)DHid * DENCC * 2);
  float*          bcat      = (float*)alloc(4 * DHid * 4);
  unsigned short* mean_bf   = (unsigned short*)alloc((long)BB * DENCC * 2);
  float*          h_f       = (float*)alloc((long)BB * DHid * 4);
  float*          c_f       = (float*)alloc((long)BB * DHid * 4);
  unsigned short* h_bf      = (unsigned short*)alloc((long)BB * DHid * 2);
  unsigned short* hnew_bf   = (unsigned short*)alloc((long)BB * DHid * 2);
  float*          dec_proj  = (float*)alloc((long)BB * AAtt * 4);
  float*          gatepre   = (float*)alloc((long)BB * DENCC * 4);
  float*          att       = (float*)alloc((long)BB * PP * 4);
  float*          alpha     = (float*)alloc((long)BB * PP * 4);
  float*          awe       = (float*)alloc((long)BB * DENCC * 4);
  unsigned short* x_bf      = (unsigned short*)alloc((long)BB * XKc * 2);
  float*          gates     = (float*)alloc((long)BB * 4 * DHid * 4);

  auto gemm = [&](const unsigned short* X, const unsigned short* W, const float* bias,
                  float* Dm, int M, int N, int K, long ldd, int mode, int t) {
    int tiles  = (M / 16) * (N / 16);
    int blocks = (tiles + 7) / 8;                 // 8 waves per 256-thread block
    k_gemm_bf16<<<blocks, 256, 0, stream>>>((const bf16_t*)X, (const bf16_t*)W, bias, Dm,
                                            M, N, K, ldd, mode, ws_declen, t);
  };

  // ---- precompute ---------------------------------------------------------
  k_sort<<<1, BB, 0, stream>>>(cap_len, enc_caps, ws_order, ws_declen,
                               out_caps, out_declen, out_order);
  k_enc_gather<<<4096, 256, 0, stream>>>(encoder_out, ws_order, enc_bf);
  k_mean<<<BB, 256, 0, stream>>>(encoder_out, ws_order, mean_bf);
  k_cvt<<<2048, 256, 0, stream>>>(W_enc_att, Wenc_bf, (long)AAtt * DENCC);
  k_cvt<<<1024, 256, 0, stream>>>(W_dec_att, Wdec_bf, (long)AAtt * DHid);
  k_cvt<<<2048, 256, 0, stream>>>(W_f_beta, Wfb_bf, (long)DENCC * DHid);
  k_cvt<<<4096, 256, 0, stream>>>(W_fc, Wfc_bf, (long)VVoc * DHid);
  k_cvt<<<2048, 256, 0, stream>>>(W_init_h, Wih0_bf, (long)DHid * DENCC);
  k_cvt<<<2048, 256, 0, stream>>>(W_init_c, Wic0_bf, (long)DHid * DENCC);
  k_wcat<<<4096, 256, 0, stream>>>(W_ih, W_hh, Wcat_bf);
  k_bcat<<<8, 256, 0, stream>>>(b_ih, b_hh, bcat);

  // enc_proj = enc @ W_enc_att^T + b_enc_att   [12544 x 512], K=2048
  gemm(enc_bf, Wenc_bf, b_enc_att, enc_proj, BB * PP, AAtt, DENCC, AAtt, 0, 0);
  // h0, c0
  gemm(mean_bf, Wih0_bf, b_init_h, h_f, BB, DHid, DENCC, DHid, 0, 0);
  gemm(mean_bf, Wic0_bf, b_init_c, c_f, BB, DHid, DENCC, DHid, 0, 0);
  k_cvt<<<128, 256, 0, stream>>>(h_f, h_bf, (long)BB * DHid);

  // ---- timestep loop ------------------------------------------------------
  for (int t = 0; t < TSt; ++t) {
    // additive attention
    gemm(h_bf, Wdec_bf, b_dec_att, dec_proj, BB, AAtt, DHid, AAtt, 0, 0);
    gemm(h_bf, Wfb_bf, b_f_beta, gatepre, BB, DENCC, DHid, DENCC, 0, 0);
    k_att<<<(BB * PP + 7) / 8, 256, 0, stream>>>(enc_proj, dec_proj, w_full, b_full, att);
    k_softmax<<<BB, 256, 0, stream>>>(att, alpha, out_alphas, ws_declen, t);
    k_awe<<<BB, 256, 0, stream>>>(alpha, enc_bf, awe);
    // LSTM input + cell
    k_buildx<<<BB, 256, 0, stream>>>(enc_caps, ws_order, emb_table, gatepre, awe, h_f, x_bf, t);
    gemm(x_bf, Wcat_bf, bcat, gates, BB, 4 * DHid, XKc, 4 * DHid, 0, 0);
    k_lstm<<<(BB * DHid + 255) / 256, 256, 0, stream>>>(gates, h_f, c_f, h_bf, hnew_bf,
                                                        ws_declen, t);
    // classifier: preds[:, t, :] = (h_new @ W_fc^T + b_fc) * active
    gemm(hnew_bf, Wfc_bf, b_fc, out_preds + (long)t * VVoc, BB, VVoc, DHid,
         (long)TSt * VVoc, 2, t);
  }
}
